// LocalTransformerBlock_44779329028681
// MI455X (gfx1250) — matrix operand order repaired and checked by reading, not compiled, not run on hardware
//
#include <hip/hip_runtime.h>
#include <math.h>

#define DIMC   512
#define HEADS  8
#define DHEAD  64
#define WINSZ  128
#define NTOK   4096
#define NBATCH 8
#define ROWS   (NBATCH * NTOK)   // 32768
#define QKSCL  8.0f
#define LNEPS  1e-5f

typedef __attribute__((ext_vector_type(16))) __bf16 bf16x16;
typedef __attribute__((ext_vector_type(8)))  __bf16 bf16x8;
typedef __attribute__((ext_vector_type(8)))  float  f32x8;

union FragU { bf16x16 v; bf16x8 h[2]; };

// ---- CDNA5 async global->LDS path (guarded; falls back to sync staging) ----
#if defined(__has_builtin)
#if __has_builtin(__builtin_amdgcn_global_load_async_to_lds_b128)
#define HAS_ASYNC_LDS 1
#endif
#endif
#ifndef HAS_ASYNC_LDS
#define HAS_ASYNC_LDS 0
#endif

// GCC-style int4 vector: matches the builtin's parameter type
// '__attribute__((__vector_size__(4*sizeof(int)))) int __device__ *'
typedef int v4i_gcc __attribute__((vector_size(16)));

__device__ inline void stage16(const __bf16* g, __bf16* l) {
#if HAS_ASYNC_LDS
  __builtin_amdgcn_global_load_async_to_lds_b128(
      (__attribute__((address_space(1))) v4i_gcc*)(void*)g,
      (__attribute__((address_space(3))) v4i_gcc*)(void*)l, 0, 0);
#else
  *(uint4*)l = *(const uint4*)g;
#endif
}

__device__ inline void stage_wait() {
#if HAS_ASYNC_LDS
#if __has_builtin(__builtin_amdgcn_s_wait_asynccnt)
  __builtin_amdgcn_s_wait_asynccnt(0);
#else
  asm volatile("s_wait_asynccnt 0" ::: "memory");
#endif
#endif
}

// D = A(16x32) * B(32x16) + C, bf16 in / f32 out
__device__ inline f32x8 wmma_bf16(bf16x16 a, bf16x16 b, f32x8 c) {
  return __builtin_amdgcn_wmma_f32_16x16x32_bf16(false, a, false, b, (short)0, c,
                                                 false, false);
}

// CDNA5 16-bit A-matrix 16x32 layout (ISA 7.12.2):
// lanes 0-15: M=lane, K = {0..7, 16..23}; lanes 16-31: M=lane-16, K = {8..15, 24..31}
__device__ inline bf16x16 load_fragA(const __bf16* base, int ldr, int lane) {
  const int r   = lane & 15;
  const int klo = (lane < 16) ? 0 : 8;
  const __bf16* p = base + r * ldr + klo;
  FragU u;
  u.h[0] = *(const bf16x8*)(p);
  u.h[1] = *(const bf16x8*)(p + 16);
  return u.v;
}

// 16-bit B-matrix 32x16 layout:
// lanes 0-15: N=lane, K = {0..7, 16..23}; lanes 16-31: N=lane-16, K = {8..15, 24..31}
__device__ inline bf16x16 load_fragB(const __bf16* base, int ldr, int lane) {
  const int r   = lane & 15;
  const int klo = (lane < 16) ? 0 : 8;
  const __bf16* p = base + r * ldr + klo;
  FragU u;
  u.h[0] = *(const bf16x8*)(p);
  u.h[1] = *(const bf16x8*)(p + 16);
  return u.v;
}

// ---------------- weight f32 -> bf16 ----------------
__global__ void cvt_f32_bf16(const float* __restrict__ in, __bf16* __restrict__ out, int n) {
  int i = blockIdx.x * 256 + threadIdx.x;
  if (i < n) out[i] = (__bf16)in[i];
}

// ---------------- layernorm: one wave per 512-col row, bf16 out ----------------
__global__ __launch_bounds__(256) void ln_rows(const float* __restrict__ X,
                                               const float* __restrict__ g,
                                               const float* __restrict__ b,
                                               __bf16* __restrict__ Y) {
  const int lane = threadIdx.x & 31;
  const int wid  = threadIdx.x >> 5;
  const int row  = blockIdx.x * 8 + wid;
  const float* xr = X + (size_t)row * DIMC;
  float vals[16];
  float s = 0.f, ss = 0.f;
#pragma unroll
  for (int i = 0; i < 16; ++i) {
    float v = xr[lane + i * 32];
    vals[i] = v; s += v; ss += v * v;
  }
#pragma unroll
  for (int m = 16; m >= 1; m >>= 1) {
    s  += __shfl_xor(s, m, 32);
    ss += __shfl_xor(ss, m, 32);
  }
  const float mean = s * (1.f / DIMC);
  const float var  = ss * (1.f / DIMC) - mean * mean;
  const float inv  = rsqrtf(var + LNEPS);
  __bf16* yr = Y + (size_t)row * DIMC;
#pragma unroll
  for (int i = 0; i < 16; ++i) {
    int c = lane + i * 32;
    yr[c] = (__bf16)((vals[i] - mean) * inv * g[c] + b[c]);
  }
}

// ------- double-buffered WMMA GEMM: C[M,N] = A[M,K] * W[N,K]^T + epilogue ------
// EPI: 0 f32 store, 1 bf16 store, 2 f32 = acc + bias[n] + resid, 3 bf16 = gelu(acc),
//      4 f32 = acc + resid
template <int EPI>
__global__ __launch_bounds__(256) void gemm_bf16_wmma(
    const __bf16* __restrict__ A, const __bf16* __restrict__ B,
    float* __restrict__ Cf, __bf16* __restrict__ Cb,
    const float* __restrict__ bias, const float* __restrict__ resid,
    int M, int N, int K) {
  constexpr int BM = 128, BN = 128, BK = 64, LDT = 64;  // 2*(16+16) KB = 64 KB LDS
  __shared__ __align__(16) __bf16 As[2][BM * LDT];
  __shared__ __align__(16) __bf16 Bs[2][BN * LDT];
  const int lane = threadIdx.x & 31;
  const int wid  = threadIdx.x >> 5;
  const int wm   = wid & 3;   // 4 waves along M (32 rows each)
  const int wn   = wid >> 2;  // 2 waves along N (64 cols each)

  f32x8 acc[2][4];
#pragma unroll
  for (int i = 0; i < 2; ++i)
#pragma unroll
    for (int j = 0; j < 4; ++j) acc[i][j] = (f32x8){0, 0, 0, 0, 0, 0, 0, 0};

  const size_t abase = (size_t)(blockIdx.y * BM) * K;
  const size_t bbase = (size_t)(blockIdx.x * BN) * K;

  // per-thread staging slots: 4 x 16B chunks per matrix per tile
  int srow[4], scol[4];
#pragma unroll
  for (int i = 0; i < 4; ++i) {
    int c = i * 256 + threadIdx.x;  // 1024 chunks
    srow[i] = c >> 3;
    scol[i] = (c & 7) * 8;
  }

  // issue tile 0
#pragma unroll
  for (int i = 0; i < 4; ++i) {
    stage16(A + abase + (size_t)srow[i] * K + scol[i], &As[0][srow[i] * LDT + scol[i]]);
    stage16(B + bbase + (size_t)srow[i] * K + scol[i], &Bs[0][srow[i] * LDT + scol[i]]);
  }

  int buf = 0;
  for (int k0 = 0; k0 < K; k0 += BK) {
    stage_wait();
    __syncthreads();
    if (k0 + BK < K) {  // issue next tile into the other buffer (async overlap)
      const int nk = k0 + BK;
#pragma unroll
      for (int i = 0; i < 4; ++i) {
        stage16(A + abase + (size_t)srow[i] * K + nk + scol[i],
                &As[buf ^ 1][srow[i] * LDT + scol[i]]);
        stage16(B + bbase + (size_t)srow[i] * K + nk + scol[i],
                &Bs[buf ^ 1][srow[i] * LDT + scol[i]]);
      }
    }
#pragma unroll
    for (int ks = 0; ks < 2; ++ks) {
      bf16x16 af[2], bfr[4];
#pragma unroll
      for (int ms = 0; ms < 2; ++ms)
        af[ms] = load_fragA(&As[buf][(wm * 32 + ms * 16) * LDT + ks * 32], LDT, lane);
#pragma unroll
      for (int ns = 0; ns < 4; ++ns)
        bfr[ns] = load_fragB(&Bs[buf][(wn * 64 + ns * 16) * LDT + ks * 32], LDT, lane);
#pragma unroll
      for (int ms = 0; ms < 2; ++ms)
#pragma unroll
        for (int ns = 0; ns < 4; ++ns)
          acc[ms][ns] = wmma_bf16(af[ms], bfr[ns], acc[ms][ns]);
    }
    buf ^= 1;
  }

  // C/D layout: VGPR r -> M = r + (lane<16 ? 0 : 8), N = lane & 15
  const int qmoff = (lane < 16) ? 0 : 8;
  const int ncol  = lane & 15;
#pragma unroll
  for (int ms = 0; ms < 2; ++ms) {
#pragma unroll
    for (int ns = 0; ns < 4; ++ns) {
#pragma unroll
      for (int r = 0; r < 8; ++r) {
        const int m = blockIdx.y * BM + wm * 32 + ms * 16 + qmoff + r;
        const int n = blockIdx.x * BN + wn * 64 + ns * 16 + ncol;
        const float v = acc[ms][ns][r];
        const size_t idx = (size_t)m * N + n;
        if (EPI == 0) {
          Cf[idx] = v;
        } else if (EPI == 1) {
          Cb[idx] = (__bf16)v;
        } else if (EPI == 2) {
          Cf[idx] = v + bias[n] + resid[idx];
        } else if (EPI == 3) {
          Cb[idx] = (__bf16)(0.5f * v * (1.f + erff(v * 0.70710678118f)));
        } else {
          Cf[idx] = v + resid[idx];
        }
      }
    }
  }
}

// ---------------- per-head L2 norm * scale + rotary, in place on qkv bf16 ------
__global__ __launch_bounds__(256) void qk_norm_rope(__bf16* __restrict__ qkv,
                                                    const float* __restrict__ qs,
                                                    const float* __restrict__ ksc) {
  const int lane = threadIdx.x & 31;
  const int wid  = threadIdx.x >> 5;
  const int task = blockIdx.x * 8 + wid;  // ROWS*HEADS tasks
  const int row  = task >> 3;
  const int h    = task & 7;
  const float pos = (float)(row & (NTOK - 1));
  // inv_freq[j] = 10000^(-j/32), lane j owns dims (j, j+32)
  const float ang = pos * __powf(10000.f, -(float)lane * (1.f / 32.f));
  const float cs = __cosf(ang), sn = __sinf(ang);
#pragma unroll
  for (int part = 0; part < 2; ++part) {  // 0 = q, 1 = k
    __bf16* t = qkv + (size_t)row * (3 * DIMC) + part * DIMC + h * DHEAD;
    const float* sc = (part == 0) ? qs : ksc;
    float a  = (float)t[lane];
    float b2 = (float)t[lane + 32];
    float sq = a * a + b2 * b2;
#pragma unroll
    for (int m = 16; m >= 1; m >>= 1) sq += __shfl_xor(sq, m, 32);
    float nrm = fmaxf(sqrtf(sq), 1e-12f);
    const float inv = 1.f / nrm;
    a  = a * inv * sc[lane];
    b2 = b2 * inv * sc[lane + 32];
    t[lane]      = (__bf16)(a * cs - b2 * sn);
    t[lane + 32] = (__bf16)(b2 * cs + a * sn);
  }
}

// ---------------- windowed attention: one block per (b, h, window) -------------
__global__ __launch_bounds__(256) void attn_win(const __bf16* __restrict__ qkv,
                                                __bf16* __restrict__ O) {
  __shared__ __align__(16) __bf16 Vt[DHEAD * 256];   // [d][key], transposed, 32 KB
  __shared__ __align__(16) __bf16 Pl[8 * 16 * 64];   // per-wave P strip, 16 KB
  const int lane = threadIdx.x & 31;
  const int wid  = threadIdx.x >> 5;
  const int w = blockIdx.x & 31;
  const int h = (blockIdx.x >> 5) & 7;
  const int b = blockIdx.x >> 8;
  const int kbase = w * WINSZ - WINSZ;  // token index of key 0 (may be < 0)

  // stage V transposed: thread t copies key row t into column t
  {
    int gn = kbase + (int)threadIdx.x;
    gn = gn < 0 ? 0 : gn;
    const __bf16* vrow =
        qkv + (size_t)(b * NTOK + gn) * (3 * DIMC) + 2 * DIMC + h * DHEAD;
#pragma unroll
    for (int ch = 0; ch < 8; ++ch) {
      bf16x8 d = *(const bf16x8*)(vrow + ch * 8);
#pragma unroll
      for (int e = 0; e < 8; ++e) Vt[(ch * 8 + e) * 256 + threadIdx.x] = d[e];
    }
  }

  // Q fragments (A layout) straight from global: wave owns 16 query rows
  const int qrow = b * NTOK + w * WINSZ + wid * 16 + (lane & 15);
  const __bf16* qp = qkv + (size_t)qrow * (3 * DIMC) + h * DHEAD;
  bf16x16 qf[2];
#pragma unroll
  for (int ksub = 0; ksub < 2; ++ksub) {
    const int klo = (lane < 16) ? 0 : 8;
    FragU u;
    u.h[0] = *(const bf16x8*)(qp + ksub * 32 + klo);
    u.h[1] = *(const bf16x8*)(qp + ksub * 32 + klo + 16);
    qf[ksub] = u.v;
  }
  __syncthreads();

  // S = Q K^T over 16 key tiles; K B-fragments read from global
  f32x8 srow[16];
#pragma unroll
  for (int nt = 0; nt < 16; ++nt) {
    f32x8 s = (f32x8){0, 0, 0, 0, 0, 0, 0, 0};
    int gn = kbase + nt * 16 + (lane & 15);
    gn = gn < 0 ? 0 : gn;
    const __bf16* kp =
        qkv + (size_t)(b * NTOK + gn) * (3 * DIMC) + DIMC + h * DHEAD;
#pragma unroll
    for (int ksub = 0; ksub < 2; ++ksub) {
      const int klo2 = (lane < 16) ? 0 : 8;
      FragU u;
      u.h[0] = *(const bf16x8*)(kp + ksub * 32 + klo2);
      u.h[1] = *(const bf16x8*)(kp + ksub * 32 + klo2 + 16);
      s = wmma_bf16(qf[ksub], u.v, s);
    }
    srow[nt] = s;
  }

  // mask + softmax; row m = r + (lane<16?0:8), col = nt*16 + (lane&15)
  const int qmoff = (lane < 16) ? 0 : 8;
  const int ncol  = lane & 15;
#pragma unroll
  for (int r = 0; r < 8; ++r) {
    const int qpos = w * WINSZ + wid * 16 + qmoff + r;
    float mx = -1e30f;
#pragma unroll
    for (int nt = 0; nt < 16; ++nt) {
      const int kpos = kbase + nt * 16 + ncol;
      float sv = srow[nt][r] * QKSCL;
      sv = (kpos >= 0 && qpos >= kpos) ? sv : -1e9f;
      srow[nt][r] = sv;
      mx = fmaxf(mx, sv);
    }
#pragma unroll
    for (int m = 8; m >= 1; m >>= 1) mx = fmaxf(mx, __shfl_xor(mx, m, 32));
    float sum = 0.f;
#pragma unroll
    for (int nt = 0; nt < 16; ++nt) {
      float p = __expf(srow[nt][r] - mx);
      srow[nt][r] = p;
      sum += p;
    }
#pragma unroll
    for (int m = 8; m >= 1; m >>= 1) sum += __shfl_xor(sum, m, 32);
    const float rinv = 1.f / sum;
#pragma unroll
    for (int nt = 0; nt < 16; ++nt) srow[nt][r] *= rinv;
  }

  // O = P V: round-trip P through per-wave LDS strip to get A-fragment layout
  __bf16* pw = &Pl[wid * 16 * 64];
  f32x8 oacc[4];
#pragma unroll
  for (int dt = 0; dt < 4; ++dt) oacc[dt] = (f32x8){0, 0, 0, 0, 0, 0, 0, 0};
#pragma unroll
  for (int pass = 0; pass < 4; ++pass) {  // 64 keys per pass
#pragma unroll
    for (int tt = 0; tt < 4; ++tt) {
      const int nt = pass * 4 + tt;
#pragma unroll
      for (int r = 0; r < 8; ++r)
        pw[(qmoff + r) * 64 + tt * 16 + ncol] = (__bf16)srow[nt][r];
    }
#pragma unroll
    for (int ksub = 0; ksub < 2; ++ksub) {
      bf16x16 af = load_fragA(pw + ksub * 32, 64, lane);
#pragma unroll
      for (int dt = 0; dt < 4; ++dt) {
        bf16x16 vf =
            load_fragB(&Vt[(dt * 16) * 256 + pass * 64 + ksub * 32], 256, lane);
        oacc[dt] = wmma_bf16(af, vf, oacc[dt]);
      }
    }
  }

  const int orow = b * NTOK + w * WINSZ + wid * 16;
#pragma unroll
  for (int dt = 0; dt < 4; ++dt)
#pragma unroll
    for (int r = 0; r < 8; ++r)
      O[(size_t)(orow + qmoff + r) * DIMC + h * DHEAD + dt * 16 + ncol] =
          (__bf16)oacc[dt][r];
}

// ---------------- host orchestration ----------------
extern "C" void kernel_launch(void* const* d_in, const int* in_sizes, int n_in,
                              void* d_out, int out_size, void* d_ws, size_t ws_size,
                              hipStream_t stream) {
  const float* x       = (const float*)d_in[0];
  const float* w_qkv   = (const float*)d_in[1];
  const float* q_scale = (const float*)d_in[2];
  const float* k_scale = (const float*)d_in[3];
  const float* w_out   = (const float*)d_in[4];
  const float* b_out   = (const float*)d_in[5];
  const float* ln1_g   = (const float*)d_in[6];
  const float* ln1_b   = (const float*)d_in[7];
  const float* ff_g    = (const float*)d_in[8];
  const float* ff_b    = (const float*)d_in[9];
  const float* w_ff1   = (const float*)d_in[10];
  const float* w_ff2   = (const float*)d_in[11];
  float* out = (float*)d_out;
  (void)in_sizes; (void)n_in; (void)out_size; (void)ws_size;

  char* ws = (char*)d_ws;
  size_t off = 0;
  auto alloc = [&](size_t bytes) -> void* {
    void* p = ws + off;
    off += (bytes + 255) & ~(size_t)255;
    return p;
  };
  __bf16* wqkv_b = (__bf16*)alloc((size_t)3 * DIMC * DIMC * 2);
  __bf16* wout_b = (__bf16*)alloc((size_t)DIMC * DIMC * 2);
  __bf16* wff1_b = (__bf16*)alloc((size_t)4 * DIMC * DIMC * 2);
  __bf16* wff2_b = (__bf16*)alloc((size_t)DIMC * 4 * DIMC * 2);
  __bf16* hbuf   = (__bf16*)alloc((size_t)ROWS * DIMC * 2);       // h / attn-out / hf
  __bf16* qkvb   = (__bf16*)alloc((size_t)ROWS * 3 * DIMC * 2);
  float*  x2     = (float*)alloc((size_t)ROWS * DIMC * 4);
  __bf16* ff1b   = (__bf16*)alloc((size_t)ROWS * 4 * DIMC * 2);

  cvt_f32_bf16<<<(3 * DIMC * DIMC + 255) / 256, 256, 0, stream>>>(w_qkv, wqkv_b, 3 * DIMC * DIMC);
  cvt_f32_bf16<<<(DIMC * DIMC + 255) / 256, 256, 0, stream>>>(w_out, wout_b, DIMC * DIMC);
  cvt_f32_bf16<<<(4 * DIMC * DIMC + 255) / 256, 256, 0, stream>>>(w_ff1, wff1_b, 4 * DIMC * DIMC);
  cvt_f32_bf16<<<(4 * DIMC * DIMC + 255) / 256, 256, 0, stream>>>(w_ff2, wff2_b, 4 * DIMC * DIMC);

  // h = LN1(x)
  ln_rows<<<ROWS / 8, 256, 0, stream>>>(x, ln1_g, ln1_b, hbuf);
  // qkv = h @ w_qkv^T  (bf16 out)
  gemm_bf16_wmma<1><<<dim3((3 * DIMC) / 128, ROWS / 128), 256, 0, stream>>>(
      hbuf, wqkv_b, nullptr, qkvb, nullptr, nullptr, ROWS, 3 * DIMC, DIMC);
  // q,k: l2norm * scale + rotary, in place
  qk_norm_rope<<<(ROWS * HEADS) / 8, 256, 0, stream>>>(qkvb, q_scale, k_scale);
  // windowed attention -> hbuf (bf16, head-major cols)
  attn_win<<<NBATCH * HEADS * (NTOK / WINSZ), 256, 0, stream>>>(qkvb, hbuf);
  // x2 = x + attn @ w_out^T + b_out  (f32)
  gemm_bf16_wmma<2><<<dim3(DIMC / 128, ROWS / 128), 256, 0, stream>>>(
      hbuf, wout_b, x2, nullptr, b_out, x, ROWS, DIMC, DIMC);
  // hf = LN2(x2)
  ln_rows<<<ROWS / 8, 256, 0, stream>>>(x2, ff_g, ff_b, hbuf);
  // ff1 = gelu(hf @ w_ff1^T)  (bf16)
  gemm_bf16_wmma<3><<<dim3((4 * DIMC) / 128, ROWS / 128), 256, 0, stream>>>(
      hbuf, wff1_b, nullptr, ff1b, nullptr, nullptr, ROWS, 4 * DIMC, DIMC);
  // out = x2 + ff1 @ w_ff2^T  (f32)
  gemm_bf16_wmma<4><<<dim3(DIMC / 128, ROWS / 128), 256, 0, stream>>>(
      ff1b, wff2_b, out, nullptr, nullptr, x2, ROWS, DIMC, 4 * DIMC);
}
